// HierarchicalAttention_6468220748527
// MI455X (gfx1250) — compile-verified, matrix operand order checked by reading, not run on hardware
//
#include <hip/hip_runtime.h>
#include <hip/hip_bf16.h>

// ---------------------------------------------------------------------------
// Hierarchical attention for MI455X (gfx1250, wave32, WMMA bf16).
//   B=4, S=8192, D=1024, H=16, hd=64, CHUNK=256, nC=32
// Pipeline:
//   x -> bf16; W* -> bf16 transposed
//   q,k,v = xbf @ W{q,k,v}_l              (WMMA GEMM)
//   ctx   = local flash-attn per (chunk,head)  (WMMA)
//   out   = ctx @ Wo_l + x                (WMMA GEMM, fp32 residual out)
//   sum   = chunk mean of out
//   qg,kg,vg = sumbf @ W{q,k,v}_g         (WMMA GEMM)
//   ctxg  = global attn per (batch,head), SEQ=32 (same WMMA kernel)
//   gctx  = ctxg @ Wo_g + sum             (WMMA GEMM, fp32 residual)
//   out  += broadcast(gctx)
// ---------------------------------------------------------------------------

typedef __bf16 bf16_t;
typedef __attribute__((ext_vector_type(8)))  __bf16 v8bf;
typedef __attribute__((ext_vector_type(16))) __bf16 v16bf;
typedef __attribute__((ext_vector_type(8)))  float  v8f;

#define D_DIM   1024
#define HEADS   16
#define HD      64
#define CHUNK_L 256
#define BATCH   4
#define SEQLEN  8192
#define NCHUNK  32            // SEQLEN / CHUNK_L
#define BT      32768         // BATCH * SEQLEN  (local tokens)
#define NG      128           // BATCH * NCHUNK  (global tokens)

__device__ __forceinline__ bf16_t cvt_bf16(float f) {
  unsigned u = __builtin_bit_cast(unsigned, f);
  unsigned r = u + 0x7FFFu + ((u >> 16) & 1u);        // round-to-nearest-even
  unsigned short h = (unsigned short)(r >> 16);
  return __builtin_bit_cast(bf16_t, h);
}

// Build a 16x32 (A) / 32x16 (B) bf16 fragment for v_wmma_f32_16x16x32_bf16.
// ISA layout: lane holds elements i<8 at K=kbase+i, i>=8 at K=16+kbase+(i-8),
// where kbase=(lane/16)*8 is folded into `p` by the caller. Two b128 reads.
__device__ __forceinline__ v16bf frag16(const bf16_t* p) {
  v8bf lo = *(const v8bf*)p;
  v8bf hi = *(const v8bf*)(p + 16);
  v16bf r;
#pragma unroll
  for (int i = 0; i < 8; ++i) { r[i] = lo[i]; r[i + 8] = hi[i]; }
  return r;
}

__device__ __forceinline__ v8f wmma_bf(v16bf a, v16bf b, v8f c) {
  return __builtin_amdgcn_wmma_f32_16x16x32_bf16(false, a, false, b,
                                                 (short)0, c, false, false);
}

// ------------------------- conversion kernels ------------------------------

__global__ __launch_bounds__(256)
void cvt_x_kernel(const float* __restrict__ x, bf16_t* __restrict__ xb) {
  long i = ((long)blockIdx.x * 256 + threadIdx.x) * 8;
  float4 f0 = *(const float4*)&x[i];
  float4 f1 = *(const float4*)&x[i + 4];
  v8bf o;
  o[0] = cvt_bf16(f0.x); o[1] = cvt_bf16(f0.y);
  o[2] = cvt_bf16(f0.z); o[3] = cvt_bf16(f0.w);
  o[4] = cvt_bf16(f1.x); o[5] = cvt_bf16(f1.y);
  o[6] = cvt_bf16(f1.z); o[7] = cvt_bf16(f1.w);
  *(v8bf*)&xb[i] = o;
}

// Wt[n*1024 + k] = bf16(W[k*1024 + n])
__global__ __launch_bounds__(256)
void cvt_wt_kernel(const float* __restrict__ W, bf16_t* __restrict__ Wt) {
  int idx = blockIdx.x * 256 + threadIdx.x;     // 1024*1024 elements
  int k = idx >> 10, n = idx & 1023;
  Wt[n * D_DIM + k] = cvt_bf16(W[idx]);
}

// ------------------------------ WMMA GEMM ----------------------------------
// C[M,N] = A[M,K] * W[K,N], with Wt = W^T (row-major [N,K], bf16).
// Block: 256 threads = 8 waves; tile 128(M) x 64(N); K-step 32.
// Wave (wm,wn) computes a 32x32 region: 2 A-frags x 2 B-frags -> 4 WMMAs
// per 8 ds_load_b128 (A/B fragment reuse).

template<bool OUT_F32>
__global__ __launch_bounds__(256)
void gemm_bf16_kernel(const bf16_t* __restrict__ A, const bf16_t* __restrict__ Wt,
                      bf16_t* __restrict__ Cb, float* __restrict__ Cf,
                      const float* __restrict__ res, int N, int K) {
  __shared__ bf16_t a_sh[128 * 32];    // 8 KB
  __shared__ bf16_t b_sh[64 * 32];     // 4 KB
  const int tid  = threadIdx.x;
  const int lane = tid & 31;
  const int wave = tid >> 5;
  const int wm = wave & 3, wn = wave >> 2;
  const long m0 = (long)blockIdx.x * 128;
  const long n0 = (long)blockIdx.y * 64;
  // staging coords: A = 512 v8bf groups (2/thread), B = 256 groups (1/thread)
  const int arow0 = tid >> 1,        acg0 = (tid & 1) * 16;     // rows 0..127
  const int brow  = tid >> 2,        bcg  = (tid & 3) * 8;      // rows 0..63
  const int lr = lane & 15;
  const int kbase = (lane >> 4) * 8;

  v8f c00 = {}, c01 = {}, c10 = {}, c11 = {};
  for (int k0 = 0; k0 < K; k0 += 32) {
    __syncthreads();
    // A tile: each thread stores 16 bf16 (two v8bf) of one row
    v8bf aa0 = *(const v8bf*)&A[(m0 + arow0) * K + k0 + acg0];
    v8bf aa1 = *(const v8bf*)&A[(m0 + arow0) * K + k0 + acg0 + 8];
    *(v8bf*)&a_sh[arow0 * 32 + acg0]     = aa0;
    *(v8bf*)&a_sh[arow0 * 32 + acg0 + 8] = aa1;
    *(v8bf*)&b_sh[brow * 32 + bcg] = *(const v8bf*)&Wt[(n0 + brow) * K + k0 + bcg];
    if (k0 + 32 < K) {
      __builtin_prefetch(&A [(m0 + arow0) * K + k0 + 32 + acg0], 0, 1);
      __builtin_prefetch(&Wt[(n0 + brow)  * K + k0 + 32 + bcg ], 0, 1);
    }
    __syncthreads();
    v16bf a0 = frag16(&a_sh[(wm * 32 + lr) * 32 + kbase]);
    v16bf a1 = frag16(&a_sh[(wm * 32 + 16 + lr) * 32 + kbase]);
    v16bf b0 = frag16(&b_sh[(wn * 32 + lr) * 32 + kbase]);
    v16bf b1 = frag16(&b_sh[(wn * 32 + 16 + lr) * 32 + kbase]);
    c00 = wmma_bf(a0, b0, c00);
    c01 = wmma_bf(a0, b1, c01);
    c10 = wmma_bf(a1, b0, c10);
    c11 = wmma_bf(a1, b1, c11);
  }
  const int rbase = (lane >> 4) * 8;
#pragma unroll
  for (int j = 0; j < 8; ++j) {
    long row0 = m0 + wm * 32 + rbase + j;
    long row1 = row0 + 16;
    long col0 = n0 + wn * 32 + lr;
    long col1 = col0 + 16;
    if (OUT_F32) {
      Cf[row0 * N + col0] = c00[j] + res[row0 * N + col0];
      Cf[row0 * N + col1] = c01[j] + res[row0 * N + col1];
      Cf[row1 * N + col0] = c10[j] + res[row1 * N + col0];
      Cf[row1 * N + col1] = c11[j] + res[row1 * N + col1];
    } else {
      Cb[row0 * N + col0] = cvt_bf16(c00[j]);
      Cb[row0 * N + col1] = cvt_bf16(c01[j]);
      Cb[row1 * N + col0] = cvt_bf16(c10[j]);
      Cb[row1 * N + col1] = cvt_bf16(c11[j]);
    }
  }
}

// --------------------------- WMMA attention --------------------------------
// One block per (group, head). SEQ tokens per group, hd=64, softmax scale 1/8.
// SEQ/16 waves, each owns 16 query rows; flash-style online softmax over
// 32-key blocks. K staged row-major in LDS, V staged transposed.

template<int SEQ>
__global__ __launch_bounds__(SEQ * 2)
void attn_kernel(const bf16_t* __restrict__ Q, const bf16_t* __restrict__ Kp,
                 const bf16_t* __restrict__ Vp, bf16_t* __restrict__ Ctx) {
  constexpr int NW = SEQ / 16;
  __shared__ bf16_t k_sh[SEQ * HD];        // [key][hd]
  __shared__ bf16_t vt_sh[HD * SEQ];       // [hd][key]
  __shared__ bf16_t p_sh[NW * 16 * 32];    // per-wave P transpose scratch

  const int tid  = threadIdx.x;
  const int lane = tid & 31;
  const int wave = tid >> 5;
  const int g = blockIdx.x >> 4;           // group
  const int h = blockIdx.x & 15;           // head
  const long base = (long)g * SEQ * D_DIM + h * HD;

  // stage K and V^T : SEQ*64/8 v8bf groups, 4 per thread
#pragma unroll
  for (int i = 0; i < 4; ++i) {
    int gi  = tid + i * (SEQ * 2);
    int row = gi >> 3;
    int cg  = (gi & 7) * 8;
    *(v8bf*)&k_sh[row * HD + cg] = *(const v8bf*)&Kp[base + (long)row * D_DIM + cg];
    v8bf vv = *(const v8bf*)&Vp[base + (long)row * D_DIM + cg];
#pragma unroll
    for (int j = 0; j < 8; ++j) vt_sh[(cg + j) * SEQ + row] = vv[j];
  }
  __syncthreads();

  const int lr = lane & 15;
  const int kbase = (lane >> 4) * 8;
  const int qrow = wave * 16 + lr;
  v16bf aQ0 = frag16(&Q[base + (long)qrow * D_DIM + 0  + kbase]);   // hd 0..31
  v16bf aQ1 = frag16(&Q[base + (long)qrow * D_DIM + 32 + kbase]);   // hd 32..63

  float m[8], l[8];
#pragma unroll
  for (int j = 0; j < 8; ++j) { m[j] = -1e30f; l[j] = 0.0f; }
  v8f o0 = {}, o1 = {}, o2 = {}, o3 = {};
  const float scale = 0.125f;              // 1/sqrt(64)
  bf16_t* pw = &p_sh[wave * 512];

  for (int kb = 0; kb < SEQ / 32; ++kb) {
    const int key0 = kb * 32;
    v8f s0 = {}, s1 = {};
    {
      v16bf b00 = frag16(&k_sh[(key0 + lr) * HD + 0  + kbase]);
      v16bf b01 = frag16(&k_sh[(key0 + lr) * HD + 32 + kbase]);
      s0 = wmma_bf(aQ0, b00, s0);
      s0 = wmma_bf(aQ1, b01, s0);
      v16bf b10 = frag16(&k_sh[(key0 + 16 + lr) * HD + 0  + kbase]);
      v16bf b11 = frag16(&k_sh[(key0 + 16 + lr) * HD + 32 + kbase]);
      s1 = wmma_bf(aQ0, b10, s1);
      s1 = wmma_bf(aQ1, b11, s1);
    }
    // online softmax over this 32-key block; lane holds col lr (s0) and lr+16
    // (s1) of rows r=(lane/16)*8+j; row reduction = 16-lane half reduction.
#pragma unroll
    for (int j = 0; j < 8; ++j) {
      float x0 = s0[j] * scale, x1 = s1[j] * scale;
      float tm = fmaxf(x0, x1);
#pragma unroll
      for (int msk = 1; msk < 16; msk <<= 1) tm = fmaxf(tm, __shfl_xor(tm, msk, 32));
      float mn = fmaxf(m[j], tm);
      float f  = __expf(m[j] - mn);
      float p0 = __expf(x0 - mn), p1 = __expf(x1 - mn);
      float rs = p0 + p1;
#pragma unroll
      for (int msk = 1; msk < 16; msk <<= 1) rs += __shfl_xor(rs, msk, 32);
      l[j] = l[j] * f + rs;
      m[j] = mn;
      o0[j] *= f; o1[j] *= f; o2[j] *= f; o3[j] *= f;
      int r = (lane >> 4) * 8 + j;
      pw[r * 32 + lr]      = cvt_bf16(p0);
      pw[r * 32 + 16 + lr] = cvt_bf16(p1);
    }
    // P re-enters as A fragment (16 q rows x 32 keys); V^T gives B fragments.
    v16bf aP  = frag16(&pw[lr * 32 + kbase]);
    v16bf bv0 = frag16(&vt_sh[(0 * 16 + lr) * SEQ + key0 + kbase]);
    v16bf bv1 = frag16(&vt_sh[(1 * 16 + lr) * SEQ + key0 + kbase]);
    v16bf bv2 = frag16(&vt_sh[(2 * 16 + lr) * SEQ + key0 + kbase]);
    v16bf bv3 = frag16(&vt_sh[(3 * 16 + lr) * SEQ + key0 + kbase]);
    o0 = wmma_bf(aP, bv0, o0);
    o1 = wmma_bf(aP, bv1, o1);
    o2 = wmma_bf(aP, bv2, o2);
    o3 = wmma_bf(aP, bv3, o3);
  }

#pragma unroll
  for (int j = 0; j < 8; ++j) {
    float inv = 1.0f / l[j];
    long row = wave * 16 + (lane >> 4) * 8 + j;
    bf16_t* cp = &Ctx[base + row * D_DIM];
    cp[0 * 16 + lr] = cvt_bf16(o0[j] * inv);
    cp[1 * 16 + lr] = cvt_bf16(o1[j] * inv);
    cp[2 * 16 + lr] = cvt_bf16(o2[j] * inv);
    cp[3 * 16 + lr] = cvt_bf16(o3[j] * inv);
  }
}

// --------------------------- small kernels ---------------------------------

__global__ __launch_bounds__(256)
void summary_kernel(const float* __restrict__ out, float* __restrict__ sumf,
                    bf16_t* __restrict__ sumb) {
  int g = blockIdx.x, t = threadIdx.x;
  const float* bp = out + (long)g * CHUNK_L * D_DIM;
#pragma unroll
  for (int i = 0; i < 4; ++i) {
    int col = t + i * 256;
    float acc = 0.0f;
    for (int r = 0; r < CHUNK_L; ++r) acc += bp[(long)r * D_DIM + col];
    acc *= (1.0f / (float)CHUNK_L);
    sumf[(long)g * D_DIM + col] = acc;
    sumb[(long)g * D_DIM + col] = cvt_bf16(acc);
  }
}

__global__ __launch_bounds__(256)
void bcast_add_kernel(float* __restrict__ out, const float* __restrict__ gctx) {
  long i = ((long)blockIdx.x * 256 + threadIdx.x) * 4;
  long row = i >> 10;            // token index
  long col = i & 1023;
  long grp = row >> 8;           // b*nC + c
  float4 a = *(float4*)&out[i];
  float4 gv = *(const float4*)&gctx[grp * D_DIM + col];
  a.x += gv.x; a.y += gv.y; a.z += gv.z; a.w += gv.w;
  *(float4*)&out[i] = a;
}

// ------------------------------ launcher -----------------------------------

extern "C" void kernel_launch(void* const* d_in, const int* in_sizes, int n_in,
                              void* d_out, int out_size, void* d_ws, size_t ws_size,
                              hipStream_t stream) {
  (void)in_sizes; (void)n_in; (void)out_size; (void)ws_size;
  const float* x = (const float*)d_in[0];
  const float* Wsrc[8] = {
    (const float*)d_in[1], (const float*)d_in[2], (const float*)d_in[3],
    (const float*)d_in[4], (const float*)d_in[5], (const float*)d_in[6],
    (const float*)d_in[7], (const float*)d_in[8]
  };
  float* out = (float*)d_out;
  char* ws = (char*)d_ws;

  // workspace carve-up (bytes)
  bf16_t* xbf = (bf16_t*)(ws);                       // 64MB; reused as ctx
  bf16_t* q   = (bf16_t*)(ws +  67108864L);
  bf16_t* k   = (bf16_t*)(ws + 134217728L);
  bf16_t* v   = (bf16_t*)(ws + 201326592L);
  bf16_t* wt[8];
  for (int i = 0; i < 8; ++i) wt[i] = (bf16_t*)(ws + 268435456L + (long)i * 2097152L);
  float*  sumf = (float*) (ws + 285212672L);         // 512KB
  bf16_t* sumb = (bf16_t*)(ws + 285736960L);         // 256KB
  bf16_t* qg   = (bf16_t*)(ws + 285999104L);
  bf16_t* kg   = (bf16_t*)(ws + 286261248L);
  bf16_t* vg   = (bf16_t*)(ws + 286523392L);
  bf16_t* ctxg = (bf16_t*)(ws + 286785536L);
  float*  gctx = (float*) (ws + 287047680L);         // 512KB

  // 1) conversions
  cvt_x_kernel<<<(BT * D_DIM) / (256 * 8), 256, 0, stream>>>(x, xbf);
  for (int i = 0; i < 8; ++i)
    cvt_wt_kernel<<<(D_DIM * D_DIM) / 256, 256, 0, stream>>>(Wsrc[i], wt[i]);

  // 2) local QKV projections (tile 128x64)
  dim3 gL(BT / 128, D_DIM / 64);
  gemm_bf16_kernel<false><<<gL, 256, 0, stream>>>(xbf, wt[0], q, nullptr, nullptr, D_DIM, D_DIM);
  gemm_bf16_kernel<false><<<gL, 256, 0, stream>>>(xbf, wt[1], k, nullptr, nullptr, D_DIM, D_DIM);
  gemm_bf16_kernel<false><<<gL, 256, 0, stream>>>(xbf, wt[2], v, nullptr, nullptr, D_DIM, D_DIM);

  // 3) local attention: ctx (reuses xbf buffer)
  bf16_t* ctx = xbf;
  attn_kernel<CHUNK_L><<<(BT / CHUNK_L) * HEADS, CHUNK_L * 2, 0, stream>>>(q, k, v, ctx);

  // 4) out = ctx @ Wo_l + x  (fp32, = "chunks")
  gemm_bf16_kernel<true><<<gL, 256, 0, stream>>>(ctx, wt[3], nullptr, out, x, D_DIM, D_DIM);

  // 5) chunk summaries
  summary_kernel<<<NG, 256, 0, stream>>>(out, sumf, sumb);

  // 6) global QKV projections (M = 128 -> one block row)
  dim3 gG(NG / 128, D_DIM / 64);
  gemm_bf16_kernel<false><<<gG, 256, 0, stream>>>(sumb, wt[4], qg, nullptr, nullptr, D_DIM, D_DIM);
  gemm_bf16_kernel<false><<<gG, 256, 0, stream>>>(sumb, wt[5], kg, nullptr, nullptr, D_DIM, D_DIM);
  gemm_bf16_kernel<false><<<gG, 256, 0, stream>>>(sumb, wt[6], vg, nullptr, nullptr, D_DIM, D_DIM);

  // 7) global attention: 4 groups of 32 summaries
  attn_kernel<32><<<BATCH * HEADS, 32 * 2, 0, stream>>>(qg, kg, vg, ctxg);

  // 8) gctx = ctxg @ Wo_g + sum  (fp32)
  gemm_bf16_kernel<true><<<gG, 256, 0, stream>>>(ctxg, wt[7], nullptr, gctx, sumf, D_DIM, D_DIM);

  // 9) out += broadcast(gctx)
  bcast_add_kernel<<<(BT * D_DIM) / (256 * 4), 256, 0, stream>>>(out, gctx);
}
